// tb_1x1_3x3dw_1x1_block_4939212390879
// MI455X (gfx1250) — 8-bit matrix operand order repaired and checked by reading, compiled, not run on hardware
//
#include <hip/hip_runtime.h>

typedef int v8i __attribute__((ext_vector_type(8)));

constexpr int CIN   = 64;
constexpr int CMID  = 384;
constexpr int COUT  = 64;
constexpr int ROWS  = 128;
constexpr int SHIFT = 16;

__device__ __forceinline__ int quant_i(int psum, int bias, int scale, int lo) {
  int t = (psum + bias) * scale + (1 << (SHIFT - 1));
  t >>= SHIFT;
  t = t < lo ? lo : t;
  t = t > 127 ? 127 : t;
  return t;
}

__device__ __forceinline__ int bs_bias(unsigned bs) {
  return (int)(short)(unsigned short)(bs & 0xffffu);
}
__device__ __forceinline__ int bs_scale(unsigned bs) {
  return (int)bs >> 16;
}

__device__ __forceinline__ int dot4_ss(unsigned a, unsigned b, int c) {
#if __has_builtin(__builtin_amdgcn_sudot4)
  return __builtin_amdgcn_sudot4(true, (int)a, true, (int)b, c, false);
#elif __has_builtin(__builtin_amdgcn_sdot4)
  return __builtin_amdgcn_sdot4((int)a, (int)b, c, false);
#else
  int r = c;
#pragma unroll
  for (int i = 0; i < 4; ++i)
    r += (int)(signed char)(a >> (8 * i)) * (int)(signed char)(b >> (8 * i));
  return r;
#endif
}

__device__ __forceinline__ void tr4(unsigned d0, unsigned d1, unsigned d2,
                                    unsigned d3, unsigned& c0, unsigned& c1,
                                    unsigned& c2, unsigned& c3) {
  unsigned lo01 = __builtin_amdgcn_perm(d1, d0, 0x05010400u);
  unsigned hi01 = __builtin_amdgcn_perm(d1, d0, 0x07030602u);
  unsigned lo23 = __builtin_amdgcn_perm(d3, d2, 0x05010400u);
  unsigned hi23 = __builtin_amdgcn_perm(d3, d2, 0x07030602u);
  c0 = __builtin_amdgcn_perm(lo23, lo01, 0x05040100u);
  c1 = __builtin_amdgcn_perm(lo23, lo01, 0x07060302u);
  c2 = __builtin_amdgcn_perm(hi23, hi01, 0x05040100u);
  c3 = __builtin_amdgcn_perm(hi23, hi01, 0x07060302u);
}

__device__ __forceinline__ v8i load_a_frag(const char* base, int stride) {
  const int lane = threadIdx.x & 31;
  const int m = lane & 15, half = lane >> 4;
  const char* p = base + m * stride + half * 16;
  v8i a;
#pragma unroll
  for (int j = 0; j < 8; ++j)
    a[j] = *(const int*)(p + (j >> 2) * 32 + (j & 3) * 4);
  return a;
}

__device__ __forceinline__ v8i load_b_frag(const char* base, int stride) {
  const int lane = threadIdx.x & 31;
  const int n = lane & 15, half = lane >> 4;
  const char* p = base + n * stride + half * 16;
  v8i b;
#pragma unroll
  for (int j = 0; j < 8; ++j)
    b[j] = *(const int*)(p + (j >> 2) * 32 + (j & 3) * 4);
  return b;
}

__global__ __launch_bounds__(256)
void k1_expand(const float* __restrict__ feats, const float* __restrict__ w1,
               const float* __restrict__ b1f, const float* __restrict__ s1f,
               char* __restrict__ x1, int N, int PAD) {
  __shared__ char     sW[CMID][CIN];
  __shared__ char     sA[ROWS][CIN];
  __shared__ unsigned sBS1[CMID];
  const int tid  = threadIdx.x;
  const int row0 = blockIdx.x * ROWS;

  for (int i = tid; i < CIN * CMID; i += 256) {
    int k = i / CMID, c = i % CMID;
    sW[c][k] = (char)(int)w1[i];
  }
  for (int i = tid; i < CMID; i += 256)
    sBS1[i] = (((unsigned)(int)s1f[i]) << 16) | (((int)b1f[i]) & 0xffff);
  for (int i = tid; i < ROWS * CIN; i += 256) {
    int r = i / CIN, k = i % CIN;
    int g = row0 + r;
    sA[r][k] = (g < N) ? (char)(int)feats[(long)g * CIN + k] : (char)0;
  }
  if (blockIdx.x == 0) {
    for (int i = tid; i < CMID; i += 256) x1[(long)PAD * CMID + i] = 0;
  }
  __syncthreads();

  const int wave = tid >> 5, lane = tid & 31;
  const int m = lane & 15, half = lane >> 4;
  const int rt = wave;
  v8i a = load_a_frag(&sA[rt * 16][0], CIN);
  const int rbase = row0 + rt * 16 + half * 8;

#pragma unroll 4
  for (int ct = 0; ct < CMID / 16; ++ct) {
    v8i b = load_b_frag(&sW[ct * 16][0], CIN);
    v8i acc = {0, 0, 0, 0, 0, 0, 0, 0};
    acc = __builtin_amdgcn_wmma_i32_16x16x64_iu8(true, a, true, b, acc,
                                                 false, false);
    const int col   = ct * 16 + m;
    const unsigned bs = sBS1[col];
    const int bias  = bs_bias(bs);
    const int scale = bs_scale(bs);
    char* p = x1 + (long)rbase * CMID + col;
#pragma unroll
    for (int j = 0; j < 8; ++j)
      p[(long)j * CMID] = (char)quant_i(acc[j], bias, scale, 0);
  }
}

__global__ __launch_bounds__(256)
void k2_dw_project(const char* __restrict__ x1, const int* __restrict__ nbr,
                   const float* __restrict__ wdw, const float* __restrict__ b2f,
                   const float* __restrict__ s2f, const float* __restrict__ w3,
                   const float* __restrict__ b3f, const float* __restrict__ s3f,
                   float* __restrict__ out, int N, int PAD) {
  __shared__ char     sX2[ROWS][CMID];
  __shared__ char     sW3[COUT][CMID];
  __shared__ unsigned sWp[2][CMID];
  __shared__ char     sW8[CMID];
  __shared__ int      sNbr[ROWS][9];
  __shared__ unsigned sBS2[CMID];
  __shared__ unsigned sBS3[COUT];

  const int tid  = threadIdx.x;
  const int row0 = blockIdx.x * ROWS;

  for (int c = tid; c < CMID; c += 256) {
    unsigned g0 = 0, g1 = 0;
#pragma unroll
    for (int k = 0; k < 4; ++k) {
      g0 |= (((unsigned)((int)wdw[k * CMID + c] & 255)) << (8 * k));
      g1 |= (((unsigned)((int)wdw[(k + 4) * CMID + c] & 255)) << (8 * k));
    }
    sWp[0][c] = g0;
    sWp[1][c] = g1;
    sW8[c] = (char)(int)wdw[8 * CMID + c];
    sBS2[c] = (((unsigned)(int)s2f[c]) << 16) | (((int)b2f[c]) & 0xffff);
  }
  if (tid < COUT)
    sBS3[tid] = (((unsigned)(int)s3f[tid]) << 16) | (((int)b3f[tid]) & 0xffff);
  for (int i = tid; i < CMID * COUT; i += 256) {
    int k = i / COUT, c = i % COUT;
    sW3[c][k] = (char)(int)w3[i];
  }
  for (int i = tid; i < ROWS * 9; i += 256) {
    int r = i / 9, k = i % 9;
    int g = row0 + r;
    int v = (g < N) ? nbr[(long)g * 9 + k] : N;
    sNbr[r][k] = (v >= N) ? PAD : v;
  }
  __syncthreads();

  for (int t = tid; t < ROWS * (CMID / 8); t += 256) {
    const int r  = t / (CMID / 8);
    const int c0 = (t % (CMID / 8)) * 8;
    uint2 d[9];
#pragma unroll
    for (int k = 0; k < 9; ++k)
      d[k] = *(const uint2*)(x1 + (long)sNbr[r][k] * CMID + c0);

    int p[8];
    unsigned a0, a1, a2, a3;
    tr4(d[0].x, d[1].x, d[2].x, d[3].x, a0, a1, a2, a3);
    p[0] = dot4_ss(a0, sWp[0][c0 + 0], 0);
    p[1] = dot4_ss(a1, sWp[0][c0 + 1], 0);
    p[2] = dot4_ss(a2, sWp[0][c0 + 2], 0);
    p[3] = dot4_ss(a3, sWp[0][c0 + 3], 0);
    tr4(d[4].x, d[5].x, d[6].x, d[7].x, a0, a1, a2, a3);
    p[0] = dot4_ss(a0, sWp[1][c0 + 0], p[0]);
    p[1] = dot4_ss(a1, sWp[1][c0 + 1], p[1]);
    p[2] = dot4_ss(a2, sWp[1][c0 + 2], p[2]);
    p[3] = dot4_ss(a3, sWp[1][c0 + 3], p[3]);
    tr4(d[0].y, d[1].y, d[2].y, d[3].y, a0, a1, a2, a3);
    p[4] = dot4_ss(a0, sWp[0][c0 + 4], 0);
    p[5] = dot4_ss(a1, sWp[0][c0 + 5], 0);
    p[6] = dot4_ss(a2, sWp[0][c0 + 6], 0);
    p[7] = dot4_ss(a3, sWp[0][c0 + 7], 0);
    tr4(d[4].y, d[5].y, d[6].y, d[7].y, a0, a1, a2, a3);
    p[4] = dot4_ss(a0, sWp[1][c0 + 4], p[4]);
    p[5] = dot4_ss(a1, sWp[1][c0 + 5], p[5]);
    p[6] = dot4_ss(a2, sWp[1][c0 + 6], p[6]);
    p[7] = dot4_ss(a3, sWp[1][c0 + 7], p[7]);
#pragma unroll
    for (int c = 0; c < 4; ++c) {
      p[c]     += (int)((d[8].x >> (8 * c)) & 255u) * (int)sW8[c0 + c];
      p[4 + c] += (int)((d[8].y >> (8 * c)) & 255u) * (int)sW8[c0 + 4 + c];
    }
    unsigned lo = 0, hi = 0;
#pragma unroll
    for (int c = 0; c < 4; ++c) {
      const unsigned bsl = sBS2[c0 + c], bsh = sBS2[c0 + 4 + c];
      lo |= (unsigned)(quant_i(p[c], bs_bias(bsl), bs_scale(bsl), 0) & 255)
            << (8 * c);
      hi |= (unsigned)(quant_i(p[4 + c], bs_bias(bsh), bs_scale(bsh), 0) & 255)
            << (8 * c);
    }
    *(uint2*)&sX2[r][c0] = make_uint2(lo, hi);
  }
  __syncthreads();

  const int wave = tid >> 5, lane = tid & 31;
  const int m = lane & 15, half = lane >> 4;
  const int rt = wave;
  const int rbase = row0 + rt * 16 + half * 8;
  const bool full = (row0 + ROWS) <= N;

  v8i acc[COUT / 16];
#pragma unroll
  for (int ct = 0; ct < COUT / 16; ++ct)
    acc[ct] = (v8i){0, 0, 0, 0, 0, 0, 0, 0};
#pragma unroll
  for (int kk = 0; kk < CMID / 64; ++kk) {
    const v8i a = load_a_frag(&sX2[rt * 16][kk * 64], CMID);
#pragma unroll
    for (int ct = 0; ct < COUT / 16; ++ct) {
      v8i b = load_b_frag(&sW3[ct * 16][kk * 64], CMID);
      acc[ct] = __builtin_amdgcn_wmma_i32_16x16x64_iu8(true, a, true, b,
                                                       acc[ct], false, false);
    }
  }
#pragma unroll
  for (int ct = 0; ct < COUT / 16; ++ct) {
    const int col   = ct * 16 + m;
    const unsigned bs = sBS3[col];
    const int bias  = bs_bias(bs);
    const int scale = bs_scale(bs);
    float* p = out + (long)rbase * COUT + col;
    if (full) {
#pragma unroll
      for (int j = 0; j < 8; ++j)
        p[(long)j * COUT] = (float)quant_i(acc[ct][j], bias, scale, -128);
    } else {
#pragma unroll
      for (int j = 0; j < 8; ++j)
        if (rbase + j < N)
          p[(long)j * COUT] = (float)quant_i(acc[ct][j], bias, scale, -128);
    }
  }
}

extern "C" void kernel_launch(void* const* d_in, const int* in_sizes, int n_in,
                              void* d_out, int out_size, void* d_ws, size_t ws_size,
                              hipStream_t stream) {
  const float* feats = (const float*)d_in[0];
  const float* w1    = (const float*)d_in[1];
  const float* b1    = (const float*)d_in[2];
  const float* s1    = (const float*)d_in[3];
  const float* wdw   = (const float*)d_in[4];
  const float* b2    = (const float*)d_in[5];
  const float* s2    = (const float*)d_in[6];
  const float* w3    = (const float*)d_in[7];
  const float* b3    = (const float*)d_in[8];
  const float* s3    = (const float*)d_in[9];
  const int*   nbr   = (const int*)d_in[10];
  float* out = (float*)d_out;

  const int N    = in_sizes[0] / CIN;
  const int nblk = (N + ROWS - 1) / ROWS;
  const int PAD  = nblk * ROWS;
  char* x1 = (char*)d_ws;

  k1_expand<<<nblk, 256, 0, stream>>>(feats, w1, b1, s1, x1, N, PAD);
  k2_dw_project<<<nblk, 256, 0, stream>>>(x1, nbr, wdw, b2, s2, w3, b3, s3,
                                          out, N, PAD);
}
